// Codebook_57913339020077
// MI455X (gfx1250) — matrix operand order repaired and checked by reading, not compiled, not run on hardware
//
#include <hip/hip_runtime.h>

// Problem constants (from reference): B=16, T=4096, D=256, K=1024
constexpr int kBT     = 65536;   // B*T rows
constexpr int kDim    = 256;     // embedding dim
constexpr int kK      = 1024;    // codebook size
constexpr int kT      = 4096;
constexpr int kBatch  = 16;
constexpr int kCbPad  = 264;     // padded LDS row stride (bf16 elements)

typedef __attribute__((ext_vector_type(16))) __bf16 v16bf;
typedef __attribute__((ext_vector_type(8)))  float  v8f;

union Frag {
  v16bf          bf;
  unsigned short us[16];
  uint4          q[2];
};

union Half8 {
  unsigned short us[8];
  uint4          q;
};

__device__ __forceinline__ unsigned short f2bf(float f) {
  unsigned int u = __float_as_uint(f);
  u += 0x7FFFu + ((u >> 16) & 1u);     // round-to-nearest-even
  return (unsigned short)(u >> 16);
}

// ---------------------------------------------------------------------------
// Prep (one wave per code row):
//   cn2h[k]  = 0.5 * |codebook[k]|^2       (fp32)
//   cb_bf[k] = bf16(codebook[k])           (row-major K x D)
// ---------------------------------------------------------------------------
__global__ void vq_prep(const float* __restrict__ cb, float* __restrict__ cn2h,
                        unsigned short* __restrict__ cb_bf) {
  const int wid = threadIdx.x >> 5, l = threadIdx.x & 31;
  const int row = blockIdx.x * 8 + wid;
  const float4* p = (const float4*)(cb + (size_t)row * kDim);
  float4 a = p[l * 2], b = p[l * 2 + 1];

  Half8 h;
  h.us[0] = f2bf(a.x); h.us[1] = f2bf(a.y); h.us[2] = f2bf(a.z); h.us[3] = f2bf(a.w);
  h.us[4] = f2bf(b.x); h.us[5] = f2bf(b.y); h.us[6] = f2bf(b.z); h.us[7] = f2bf(b.w);
  *(uint4*)(cb_bf + (size_t)row * kDim + l * 8) = h.q;

  float s = a.x*a.x + a.y*a.y + a.z*a.z + a.w*a.w
          + b.x*b.x + b.y*b.y + b.z*b.z + b.w*b.w;
  #pragma unroll
  for (int m = 16; m >= 1; m >>= 1) s += __shfl_xor(s, m, 32);
  if (l == 0) cn2h[row] = 0.5f * s;
}

__global__ void vq_init(float* __restrict__ acc) {
  if ((int)threadIdx.x < kBatch) acc[threadIdx.x] = 0.0f;
}

// ---------------------------------------------------------------------------
// Pass 1: argmax over k of (z . c_k - 0.5|c_k|^2) via BF16 WMMA.
// Block = 256 threads = 8 waves; wave w owns rows [blk*128 + w*16, +16).
// A fragments (16x32 bf16, 8 covering D=256) cached in VGPRs.
// bf16 codebook chunk (16 codes, 8.25 KB) double-buffered through LDS:
// one barrier per chunk, staging of chunk n+1 overlaps WMMAs of chunk n,
// prefetch of chunk n+2 issued ahead.
// ---------------------------------------------------------------------------
__global__ void __launch_bounds__(256) vq_argmin(
    const float* __restrict__ z_e, const unsigned short* __restrict__ cb_bf,
    const float* __restrict__ cn2h, int* __restrict__ code_i,
    float* __restrict__ code_f) {
  __shared__ float          s_cn2[kK];
  __shared__ unsigned short s_cb[2][16 * kCbPad];

  const int tid = threadIdx.x;
  const int wid = tid >> 5, l = tid & 31;
  const int g = l >> 4, ln = l & 15;
  const int tile_row = blockIdx.x * 128 + wid * 16;

  // Per-thread staging coordinates: code row (0..15), 16-dim slice.
  const int s_cr = tid >> 4;
  const int s_cc = (tid & 15) * 16;

  // stage 0.5*|c|^2 for all K codes
  #pragma unroll
  for (int i = 0; i < 4; i++) s_cn2[i * 256 + tid] = cn2h[i * 256 + tid];

  // Load + convert A fragments for this wave's 16 rows.
  // A (16-bit, 16x32): lane (g,ln): elems 0-7 <- K=8g..+7, elems 8-15 <- K=16+8g..+7
  Frag A[8];
  {
    const float* zr = z_e + (size_t)(tile_row + ln) * kDim;
    #pragma unroll
    for (int f = 0; f < 8; f++) {
      const int c0 = 32 * f + 8 * g;
      const float4* p0 = (const float4*)(zr + c0);
      const float4* p1 = (const float4*)(zr + c0 + 16);
      float4 x0 = p0[0], x1 = p0[1], y0 = p1[0], y1 = p1[1];
      A[f].us[0]  = f2bf(x0.x); A[f].us[1]  = f2bf(x0.y);
      A[f].us[2]  = f2bf(x0.z); A[f].us[3]  = f2bf(x0.w);
      A[f].us[4]  = f2bf(x1.x); A[f].us[5]  = f2bf(x1.y);
      A[f].us[6]  = f2bf(x1.z); A[f].us[7]  = f2bf(x1.w);
      A[f].us[8]  = f2bf(y0.x); A[f].us[9]  = f2bf(y0.y);
      A[f].us[10] = f2bf(y0.z); A[f].us[11] = f2bf(y0.w);
      A[f].us[12] = f2bf(y1.x); A[f].us[13] = f2bf(y1.y);
      A[f].us[14] = f2bf(y1.z); A[f].us[15] = f2bf(y1.w);
    }
  }

  float best_v[8];
  int   best_i[8];
  #pragma unroll
  for (int r = 0; r < 8; r++) { best_v[r] = -3.4e38f; best_i[r] = 0; }

  // Stage chunk 0 into buffer 0.
  {
    uint4 v = *(const uint4*)(cb_bf + (size_t)s_cr * kDim + s_cc);
    *(uint4*)&s_cb[0][s_cr * kCbPad + s_cc] = v;
  }
  __syncthreads();

  for (int chunk = 0; chunk < 64; chunk++) {
    const int cur = chunk & 1;

    // Stage next chunk into the other buffer (overlaps WMMAs below).
    if (chunk + 1 < 64) {
      uint4 v = *(const uint4*)(cb_bf +
                 (size_t)((chunk + 1) * 16 + s_cr) * kDim + s_cc);
      *(uint4*)&s_cb[cur ^ 1][s_cr * kCbPad + s_cc] = v;
    }
    // Prefetch chunk n+2 toward L1/L2.
    if (chunk + 2 < 64) {
      __builtin_prefetch(cb_bf + (size_t)((chunk + 2) * 16 + s_cr) * kDim + s_cc,
                         0, 1);
    }

    const float cn = s_cn2[chunk * 16 + ln];
    v8f acc;
    #pragma unroll
    for (int r = 0; r < 8; r++) acc[r] = -cn;   // seed with -0.5|c_n|^2

    const unsigned short* cbuf = s_cb[cur];
    #pragma unroll
    for (int f = 0; f < 8; f++) {
      // B (16-bit, 32x16): lane (g,ln) col ln: elems 0-7 <- K=8g..+7, elems 8-15 <- K=16+8g..+7
      Frag Bf;
      const uint4* bp = (const uint4*)&cbuf[ln * kCbPad + 32 * f + 8 * g];
      Bf.q[0] = bp[0]; Bf.q[1] = bp[2];
      acc = __builtin_amdgcn_wmma_f32_16x16x32_bf16(
          false, A[f].bf, false, Bf.bf, (short)0, acc, false, false);
    }

    const int nbase = chunk * 16 + ln;
    #pragma unroll
    for (int r = 0; r < 8; r++) {
      bool better = acc[r] > best_v[r];
      best_v[r] = better ? acc[r] : best_v[r];
      best_i[r] = better ? nbase  : best_i[r];
    }
    __syncthreads();   // next-buffer staged for all; current buffer free
  }

  // Per-row reduction across the 16 lanes of each half-wave.
  // D layout: lane l, elem r -> row (r + 8g), code class ln.
  #pragma unroll
  for (int r = 0; r < 8; r++) {
    float v = best_v[r];
    int   bi = best_i[r];
    #pragma unroll
    for (int m = 1; m < 16; m <<= 1) {
      float ov = __shfl_xor(v, m, 16);
      int   oi = __shfl_xor(bi, m, 16);
      bool take = (ov > v) || (ov == v && oi < bi);  // argmin-first tie-break
      v  = take ? ov : v;
      bi = take ? oi : bi;
    }
    if (ln == 0) {
      const int row = tile_row + r + 8 * g;
      code_i[row] = bi;
      code_f[row] = (float)bi;
    }
  }
}

// ---------------------------------------------------------------------------
// Pass 2: z_q = codebook[code] (== z_q_st forward value); exact fp32
// per-batch sum of |z_e - z_q|^2. One wave per row; one atomic per block.
// ---------------------------------------------------------------------------
__global__ void __launch_bounds__(256) vq_gather(
    const float* __restrict__ z_e, const float* __restrict__ cb,
    const int* __restrict__ code_i, float* __restrict__ z_q,
    float* __restrict__ lacc) {
  __shared__ float bsum;
  const int tid = threadIdx.x;
  const int l = tid & 31;
  const int row = blockIdx.x * 8 + (tid >> 5);
  const int k = code_i[row];

  const float4* zp = (const float4*)(z_e + (size_t)row * kDim);
  const float4* cp = (const float4*)(cb  + (size_t)k   * kDim);
  float4*       op = (float4*)(z_q + (size_t)row * kDim);

  float4 c0 = cp[l * 2], c1 = cp[l * 2 + 1];
  float4 z0 = zp[l * 2], z1 = zp[l * 2 + 1];
  op[l * 2] = c0; op[l * 2 + 1] = c1;

  float s = 0.0f, d;
  d = z0.x - c0.x; s += d * d;  d = z0.y - c0.y; s += d * d;
  d = z0.z - c0.z; s += d * d;  d = z0.w - c0.w; s += d * d;
  d = z1.x - c1.x; s += d * d;  d = z1.y - c1.y; s += d * d;
  d = z1.z - c1.z; s += d * d;  d = z1.w - c1.w; s += d * d;
  #pragma unroll
  for (int m = 16; m >= 1; m >>= 1) s += __shfl_xor(s, m, 32);

  if (tid == 0) bsum = 0.0f;
  __syncthreads();
  if (l == 0) atomicAdd(&bsum, s);
  __syncthreads();
  // 512 blocks per batch (4096 rows / 8 rows per block); blocks never straddle
  if (tid == 0) atomicAdd(&lacc[blockIdx.x >> 9], bsum);
}

__global__ void vq_final(const float* __restrict__ lacc,
                         float* __restrict__ out_losses) {
  const int t = threadIdx.x;
  if (t < kBatch) {
    float v = lacc[t] * (1.0f / ((float)kT * (float)kDim));
    out_losses[t]          = v;   // codebook_loss
    out_losses[kBatch + t] = v;   // commitment_loss (numerically identical)
  }
}

// ---------------------------------------------------------------------------
extern "C" void kernel_launch(void* const* d_in, const int* in_sizes, int n_in,
                              void* d_out, int out_size, void* d_ws, size_t ws_size,
                              hipStream_t stream) {
  const float* z_e = (const float*)d_in[0];   // (16,4096,256) fp32
  const float* cb  = (const float*)d_in[1];   // (1024,256) fp32

  float* out        = (float*)d_out;
  float* out_zq     = out;                               // 16*4096*256
  float* out_code   = out + (size_t)kBT * kDim;          // 16*4096
  float* out_losses = out_code + kBT;                    // 16 + 16

  char*           ws     = (char*)d_ws;
  float*          cn2h   = (float*)ws;                   // 1024 f32   (4 KB)
  int*            code_i = (int*)(ws + 4096);            // 65536 i32  (256 KB)
  float*          lacc   = (float*)(ws + 4096 + 262144); // 16 f32
  unsigned short* cb_bf  = (unsigned short*)(ws + 266304); // 1024*256 bf16 (512 KB)

  vq_init  <<<1, 32, 0, stream>>>(lacc);
  vq_prep  <<<kK / 8, 256, 0, stream>>>(cb, cn2h, cb_bf);
  vq_argmin<<<kBT / 128, 256, 0, stream>>>(z_e, cb_bf, cn2h, code_i, out_code);
  vq_gather<<<kBT / 8, 256, 0, stream>>>(z_e, cb, code_i, out_zq, lacc);
  vq_final <<<1, 32, 0, stream>>>(lacc, out_losses);
}
